// MultiHeadAttention_5025111736944
// MI455X (gfx1250) — matrix operand order repaired and checked by reading, not compiled, not run on hardware
//
#include <hip/hip_runtime.h>
#include <hip/hip_bf16.h>

// ---------------------------------------------------------------------------
// MHA forward for gfx1250 (MI455X): bf16 WMMA + TDM (tensor_load_to_lds)
// double-buffered B staging, flash-style softmax attention.
// Shapes: B=8, S=1024, D=1024, H=16, hd=64, 3D=3072.
// ---------------------------------------------------------------------------

typedef __attribute__((ext_vector_type(16))) __bf16 v16bf;
typedef __attribute__((ext_vector_type(8)))  __bf16 v8bf;
typedef __attribute__((ext_vector_type(8)))  float  v8f;
typedef __attribute__((ext_vector_type(4)))  unsigned int u32x4;
typedef __attribute__((ext_vector_type(4)))  int  i32x4;
typedef __attribute__((ext_vector_type(8)))  int  i32x8;

#define BB   8
#define SS   1024
#define DD   1024
#define HH   16
#define HD   64
#define D3   3072
#define MM   (BB * SS)        // 8192 rows

#if defined(__has_builtin)
#if __has_builtin(__builtin_amdgcn_tensor_load_to_lds)
#define HAVE_TDM 1
#endif
#endif
#ifndef HAVE_TDM
#define HAVE_TDM 0
#endif

// Load a 16-element bf16 WMMA operand from two contiguous 16-byte chunks.
__device__ __forceinline__ v16bf load_op(const __bf16* p0, const __bf16* p1) {
    v8bf lo = *reinterpret_cast<const v8bf*>(p0);
    v8bf hi = *reinterpret_cast<const v8bf*>(p1);
    v16bf r;
#pragma unroll
    for (int i = 0; i < 8; ++i) { r[i] = lo[i]; r[i + 8] = hi[i]; }
    return r;
}

__device__ __forceinline__ v8f bwmma(v16bf a, v16bf b, v8f c) {
    return __builtin_amdgcn_wmma_f32_16x16x32_bf16(
        /*neg_a=*/false, a, /*neg_b=*/false, b,
        /*c_mod=*/(short)0, c, /*reuse_a=*/false, /*reuse_b=*/false);
}

__device__ __forceinline__ float half_max16(float v) {
#pragma unroll
    for (int m = 1; m <= 8; m <<= 1) v = fmaxf(v, __shfl_xor(v, m, 32));
    return v;
}
__device__ __forceinline__ float half_sum16(float v) {
#pragma unroll
    for (int m = 1; m <= 8; m <<= 1) v += __shfl_xor(v, m, 32);
    return v;
}

#if HAVE_TDM
// ---------------------------------------------------------------------------
// TDM: DMA a 2-D tile (256 rows x 32 bf16, row stride DD elems) into LDS.
// D# built per CDNA5 ISA 8.3/8.4: group0 {count,lds_addr,global_addr,type=2},
// group1 {data_size=2B, tensor dims, tile dims 32x256, dim0 stride}.
// ---------------------------------------------------------------------------
__device__ __forceinline__ void tdm_load_b_tile(const __bf16* gsrc,
                                                unsigned lds_byte_off) {
    const unsigned long long ga = (unsigned long long)(uintptr_t)gsrc;
    u32x4 g0;
    g0[0] = 1u;                                    // count=1 (valid), user mode
    g0[1] = lds_byte_off;                          // D#.lds_addr  [63:32]
    g0[2] = (unsigned)ga;                          // global_addr[31:0]
    g0[3] = (unsigned)((ga >> 32) & 0x01ffffffu)   // global_addr[56:32]
            | 0x80000000u;                         // type=2 ("image") [127:126]
    i32x8 g1;
    g1[0] = 1 << 16;                               // data_size=1 -> 2 bytes/elem
    g1[1] = (int)(DD << 16);                       // tensor_dim0[15:0] = 1024
    g1[2] = (int)(256 << 16);                      // dim0 hi=0 | tensor_dim1 lo = 256
    g1[3] = (int)(32 << 16);                       // dim1 hi=0 | tile_dim0 = 32
    g1[4] = 256;                                   // tile_dim1=256, tile_dim2=0
    g1[5] = DD;                                    // tensor_dim0_stride lo32 = 1024
    g1[6] = 0;                                     // stride hi | dim1_stride lo
    g1[7] = 0;
    const i32x4 z4 = {0, 0, 0, 0};
#if __clang_major__ >= 23
    const i32x8 z8 = {0, 0, 0, 0, 0, 0, 0, 0};
    __builtin_amdgcn_tensor_load_to_lds(g0, g1, z4, z4, z8, 0);
#else
    __builtin_amdgcn_tensor_load_to_lds(g0, g1, z4, z4, 0);
#endif
}
#endif

// ---------------------------------------------------------------------------
// Shared GEMM main loop: C(32x256 per block) += A(.,K) x Wt(N,K)^T.
// B tile double-buffered in LDS via TDM; A operand software-pipelined in VGPRs.
// All four B operands are pulled from LDS before the WMMA burst so DS waits
// can be partial instead of dscnt==0 per WMMA.
// ---------------------------------------------------------------------------
__device__ __forceinline__ void gemm_mainloop(const __bf16* __restrict__ arow,
                                              const __bf16* __restrict__ WtBlock,
                                              __bf16 (&sB)[2][256][32],
                                              int wn, int l15, int kbA, int koB,
                                              v8f acc[4]) {
    const int nsteps = DD / 32;
    // ---- prologue: fill buffer 0 with k=0 tile -------------------------
#if HAVE_TDM
    if (threadIdx.x < 32) tdm_load_b_tile(WtBlock, (unsigned)(uintptr_t)&sB[0][0][0]);
#else
    {
        const uint4* s4 = reinterpret_cast<const uint4*>(WtBlock + (size_t)threadIdx.x * DD);
        uint4* dst = reinterpret_cast<uint4*>(&sB[0][threadIdx.x][0]);
#pragma unroll
        for (int j = 0; j < 4; ++j) dst[j] = s4[j];
    }
#endif
    v16bf a_cur = load_op(arow + kbA, arow + kbA + 8);

    for (int i = 0; i < nsteps; ++i) {
        const int k = i * 32;
#if HAVE_TDM
        if (threadIdx.x < 32) __builtin_amdgcn_s_wait_tensorcnt(0);
#endif
        __syncthreads();   // current buffer ready; prev reads of next buffer done
        if (i + 1 < nsteps) {
#if HAVE_TDM
            if (threadIdx.x < 32)
                tdm_load_b_tile(WtBlock + k + 32,
                                (unsigned)(uintptr_t)&sB[(i + 1) & 1][0][0]);
#else
            const uint4* s4 = reinterpret_cast<const uint4*>(
                WtBlock + (size_t)threadIdx.x * DD + k + 32);
            uint4* dst = reinterpret_cast<uint4*>(&sB[(i + 1) & 1][threadIdx.x][0]);
#pragma unroll
            for (int j = 0; j < 4; ++j) dst[j] = s4[j];
#endif
        }
        // pipeline next A operand while WMMAs consume the current one
        v16bf a_nxt = a_cur;
        if (i + 1 < nsteps) {
            const __bf16* ap = arow + k + 32 + kbA;
            __builtin_prefetch(ap + 64, 0, 1);
            a_nxt = load_op(ap, ap + 8);
        }
        const __bf16(*bb)[32] = sB[i & 1];
        const __bf16* bp0 = &bb[wn * 64 +  0 + l15][koB];
        const __bf16* bp1 = &bb[wn * 64 + 16 + l15][koB];
        const __bf16* bp2 = &bb[wn * 64 + 32 + l15][koB];
        const __bf16* bp3 = &bb[wn * 64 + 48 + l15][koB];
        const v16bf b0 = load_op(bp0, bp0 + 8);
        const v16bf b1 = load_op(bp1, bp1 + 8);
        const v16bf b2 = load_op(bp2, bp2 + 8);
        const v16bf b3 = load_op(bp3, bp3 + 8);
        acc[0] = bwmma(a_cur, b0, acc[0]);
        acc[1] = bwmma(a_cur, b1, acc[1]);
        acc[2] = bwmma(a_cur, b2, acc[2]);
        acc[3] = bwmma(a_cur, b3, acc[3]);
        a_cur = a_nxt;
    }
}

// ---------------------------------------------------------------------------
// f32 -> bf16 elementwise (vectorized x4)
// ---------------------------------------------------------------------------
__global__ __launch_bounds__(256) void cvt_bf16_x4(const float* __restrict__ in,
                                                   __bf16* __restrict__ out, int n4) {
    int i = blockIdx.x * 256 + threadIdx.x;
    if (i < n4) {
        float4 v = reinterpret_cast<const float4*>(in)[i];
        __bf16* o = out + (size_t)i * 4;
        o[0] = (__bf16)v.x; o[1] = (__bf16)v.y; o[2] = (__bf16)v.z; o[3] = (__bf16)v.w;
    }
}

// f32 [K,N] row-major -> bf16 [N,K] row-major (transpose+convert)
__global__ __launch_bounds__(256) void cvt_transpose(const float* __restrict__ W,
                                                     __bf16* __restrict__ WT,
                                                     int K, int N) {
    int idx = blockIdx.x * 256 + threadIdx.x;
    if (idx < K * N) {
        int k = idx / N, n = idx - k * N;
        WT[(size_t)n * K + k] = (__bf16)W[idx];
    }
}

// ---------------------------------------------------------------------------
// GEMM1: qkv = x @ Wc + bc -> Q[B,H,S,hd], K[B,H,S,hd], Vt[B,H,hd,S] (bf16)
// ---------------------------------------------------------------------------
__global__ __launch_bounds__(256) void gemm_qkv(const __bf16* __restrict__ A,
                                                const __bf16* __restrict__ Wt,
                                                const float* __restrict__ bias,
                                                __bf16* __restrict__ Qb,
                                                __bf16* __restrict__ Kb,
                                                __bf16* __restrict__ Vt) {
    __shared__ __bf16 sB[2][256][32];
    const int lane = threadIdx.x & 31;
    const int w    = threadIdx.x >> 5;
    const int wn   = w & 3;
    const int mbase = blockIdx.x * 32 + (w >> 2) * 16;
    const int nbase = blockIdx.y * 256 + wn * 64;
    const int l15 = lane & 15;
    const int hi8 = (lane & 16) ? 8 : 0;
    const int kbA = (lane & 16) ? 16 : 0;
    const int koB = (lane & 16) ? 16 : 0;

    const __bf16* arow    = A  + (size_t)(mbase + l15) * DD;
    const __bf16* WtBlock = Wt + (size_t)blockIdx.y * 256 * DD;
    v8f acc[4] = {{}, {}, {}, {}};
    gemm_mainloop(arow, WtBlock, sB, wn, l15, kbA, koB, acc);

#pragma unroll
    for (int t = 0; t < 4; ++t) {
        const int ng = nbase + t * 16 + l15;        // column in [0,3072)
        const float bv = bias[ng];
        const int region = ng >> 10;                // 0=Q 1=K 2=V
        const int nl = ng & 1023;
        const int h = nl >> 6, d = nl & 63;
#pragma unroll
        for (int r = 0; r < 8; ++r) {
            const int m = mbase + r + hi8;          // row in [0,8192)
            const int bidx = m >> 10, s = m & 1023;
            const __bf16 v = (__bf16)(acc[t][r] + bv);
            if (region == 2) {
                Vt[(((size_t)bidx * HH + h) * HD + d) * SS + s] = v;
            } else {
                const size_t o = (((size_t)bidx * HH + h) * SS + s) * HD + d;
                if (region == 0) Qb[o] = v; else Kb[o] = v;
            }
        }
    }
}

// ---------------------------------------------------------------------------
// Flash attention: one wave per (b,h,q-tile of 16). Anti-causal mask: keep k>=q.
// ---------------------------------------------------------------------------
__global__ __launch_bounds__(256) void attn_flash(const __bf16* __restrict__ Qb,
                                                  const __bf16* __restrict__ Kb,
                                                  const __bf16* __restrict__ Vt,
                                                  __bf16* __restrict__ Cc) {
    __shared__ __bf16 smem[8][16][32];              // per-wave 16x32 prob tile
    const int lane = threadIdx.x & 31;
    const int w    = threadIdx.x >> 5;
    const int gw   = blockIdx.x * 8 + w;            // 0..8191
    const int qt   = gw & 63;                       // S/16 = 64 tiles
    const int bh   = gw >> 6;                       // 0..127
    const int b    = bh >> 4, h = bh & 15;
    const int q0   = qt << 4;

    const int l15 = lane & 15;
    const int hi8 = (lane & 16) ? 8 : 0;
    const int kbA = (lane & 16) ? 16 : 0;
    const int koB = (lane & 16) ? 16 : 0;

    const __bf16* Qbase = Qb + (size_t)bh * SS * HD;
    const __bf16* Kbase = Kb + (size_t)bh * SS * HD;
    const __bf16* Vbase = Vt + (size_t)bh * HD * SS;

    const __bf16* qrow = Qbase + (size_t)(q0 + l15) * HD;
    const v16bf aQ0 = load_op(qrow + kbA,      qrow + kbA + 8);
    const v16bf aQ1 = load_op(qrow + 32 + kbA, qrow + 32 + kbA + 8);

    v8f acc[4] = {{}, {}, {}, {}};
    float mrow[8], lrow[8];
#pragma unroll
    for (int r = 0; r < 8; ++r) { mrow[r] = -1e30f; lrow[r] = 0.0f; }

    const float scale = 0.03125f;                   // 1/sqrt(D) with D=1024

    for (int kc = (q0 & ~31); kc < SS; kc += 32) {
        if (kc + 32 < SS) {
            __builtin_prefetch(Kbase + (size_t)(kc + 32 + l15) * HD, 0, 1);
            __builtin_prefetch(Vbase + (size_t)l15 * SS + kc + 32, 0, 1);
        }
        // ---- scores: two 16-key subtiles; batch all K-operand loads ----
        const __bf16* kp  = Kbase + (size_t)(kc + l15) * HD + koB;
        const __bf16* kp2 = Kbase + (size_t)(kc + 16 + l15) * HD + koB;
        const v16bf bK00 = load_op(kp,       kp + 8);
        const v16bf bK01 = load_op(kp + 32,  kp + 40);
        const v16bf bK10 = load_op(kp2,      kp2 + 8);
        const v16bf bK11 = load_op(kp2 + 32, kp2 + 40);
        v8f s0 = {}, s1 = {};
        s0 = bwmma(aQ0, bK00, s0);
        s0 = bwmma(aQ1, bK01, s0);
        s1 = bwmma(aQ0, bK10, s1);
        s1 = bwmma(aQ1, bK11, s1);
        // ---- scale + anti-causal mask (keep k >= q) --------------------
        const int col0 = kc + l15, col1 = kc + 16 + l15;
#pragma unroll
        for (int r = 0; r < 8; ++r) {
            const int q = q0 + r + hi8;
            float a0 = s0[r] * scale, a1 = s1[r] * scale;
            if (col0 < q) a0 = -1e30f;
            if (col1 < q) a1 = -1e30f;
            s0[r] = a0; s1[r] = a1;
        }
        // ---- online softmax (rows live per 16-lane half in C layout) ---
        float e0[8], e1[8];
#pragma unroll
        for (int r = 0; r < 8; ++r) {
            const float rmax = half_max16(fmaxf(s0[r], s1[r]));
            const float newm = fmaxf(mrow[r], rmax);
            const float alpha = __expf(mrow[r] - newm);
            e0[r] = __expf(s0[r] - newm);
            e1[r] = __expf(s1[r] - newm);
            lrow[r] = lrow[r] * alpha + half_sum16(e0[r] + e1[r]);
            mrow[r] = newm;
#pragma unroll
            for (int t = 0; t < 4; ++t) acc[t][r] *= alpha;
        }
        // ---- C-layout -> A-layout via per-wave LDS tile ----------------
#pragma unroll
        for (int r = 0; r < 8; ++r) {
            smem[w][r + hi8][l15]      = (__bf16)e0[r];
            smem[w][r + hi8][l15 + 16] = (__bf16)e1[r];
        }
        asm volatile("s_wait_dscnt 0" ::: "memory");
        const __bf16* lrowp = &smem[w][l15][0];
        const v16bf pA = load_op(lrowp + kbA, lrowp + kbA + 8);
        // ---- PV: acc[t] += P(16x32) x V(32x16); batch V-operand loads --
        const __bf16* vp0 = Vbase + (size_t)( 0 + l15) * SS + kc + koB;
        const __bf16* vp1 = Vbase + (size_t)(16 + l15) * SS + kc + koB;
        const __bf16* vp2 = Vbase + (size_t)(32 + l15) * SS + kc + koB;
        const __bf16* vp3 = Vbase + (size_t)(48 + l15) * SS + kc + koB;
        const v16bf bV0 = load_op(vp0, vp0 + 8);
        const v16bf bV1 = load_op(vp1, vp1 + 8);
        const v16bf bV2 = load_op(vp2, vp2 + 8);
        const v16bf bV3 = load_op(vp3, vp3 + 8);
        acc[0] = bwmma(pA, bV0, acc[0]);
        acc[1] = bwmma(pA, bV1, acc[1]);
        acc[2] = bwmma(pA, bV2, acc[2]);
        acc[3] = bwmma(pA, bV3, acc[3]);
    }

    // ---- normalize + store bf16 concat [B,S,D] -------------------------
#pragma unroll
    for (int r = 0; r < 8; ++r) {
        const float inv = 1.0f / lrow[r];
        const int sq = q0 + r + hi8;
#pragma unroll
        for (int t = 0; t < 4; ++t) {
            Cc[((size_t)b * SS + sq) * DD + h * 64 + t * 16 + l15] =
                (__bf16)(acc[t][r] * inv);
        }
    }
}

// ---------------------------------------------------------------------------
// GEMM2: out = concat @ Wo + bo  (f32 output)
// ---------------------------------------------------------------------------
__global__ __launch_bounds__(256) void gemm_out(const __bf16* __restrict__ A,
                                                const __bf16* __restrict__ Wt,
                                                const float* __restrict__ bias,
                                                float* __restrict__ out) {
    __shared__ __bf16 sB[2][256][32];
    const int lane = threadIdx.x & 31;
    const int w    = threadIdx.x >> 5;
    const int wn   = w & 3;
    const int mbase = blockIdx.x * 32 + (w >> 2) * 16;
    const int nbase = blockIdx.y * 256 + wn * 64;
    const int l15 = lane & 15;
    const int hi8 = (lane & 16) ? 8 : 0;
    const int kbA = (lane & 16) ? 16 : 0;
    const int koB = (lane & 16) ? 16 : 0;

    const __bf16* arow    = A  + (size_t)(mbase + l15) * DD;
    const __bf16* WtBlock = Wt + (size_t)blockIdx.y * 256 * DD;
    v8f acc[4] = {{}, {}, {}, {}};
    gemm_mainloop(arow, WtBlock, sB, wn, l15, kbA, koB, acc);

#pragma unroll
    for (int t = 0; t < 4; ++t) {
        const int ng = nbase + t * 16 + l15;
        const float bv = bias[ng];
#pragma unroll
        for (int r = 0; r < 8; ++r) {
            const int m = mbase + r + hi8;
            out[(size_t)m * DD + ng] = acc[t][r] + bv;
        }
    }
}

// ---------------------------------------------------------------------------
// Host launcher
// ---------------------------------------------------------------------------
extern "C" void kernel_launch(void* const* d_in, const int* in_sizes, int n_in,
                              void* d_out, int out_size, void* d_ws, size_t ws_size,
                              hipStream_t stream) {
    const float* x  = (const float*)d_in[0];
    const float* Wc = (const float*)d_in[1];
    const float* bc = (const float*)d_in[2];
    const float* Wo = (const float*)d_in[3];
    const float* bo = (const float*)d_in[4];
    float* out = (float*)d_out;

    char* ws = (char*)d_ws;
    size_t off = 0;
    auto carve = [&](size_t bytes) {
        char* p = ws + off;
        off += (bytes + 255) & ~(size_t)255;
        return p;
    };
    __bf16* xb  = (__bf16*)carve((size_t)MM * DD * 2);           // 16 MB
    __bf16* WcT = (__bf16*)carve((size_t)D3 * DD * 2);           //  6 MB
    __bf16* WoT = (__bf16*)carve((size_t)DD * DD * 2);           //  2 MB
    __bf16* Qb  = (__bf16*)carve((size_t)BB * HH * SS * HD * 2); // 16 MB
    __bf16* Kb  = (__bf16*)carve((size_t)BB * HH * SS * HD * 2); // 16 MB
    __bf16* Vt  = (__bf16*)carve((size_t)BB * HH * HD * SS * 2); // 16 MB
    __bf16* Cc  = (__bf16*)carve((size_t)MM * DD * 2);           // 16 MB

    // 1) precision conversions
    cvt_bf16_x4<<<(MM * DD / 4 + 255) / 256, 256, 0, stream>>>(x, xb, MM * DD / 4);
    cvt_transpose<<<(DD * D3 + 255) / 256, 256, 0, stream>>>(Wc, WcT, DD, D3);
    cvt_transpose<<<(DD * DD + 255) / 256, 256, 0, stream>>>(Wo, WoT, DD, DD);

    // 2) QKV projection (M=8192, N=3072)
    gemm_qkv<<<dim3(MM / 32, D3 / 256), 256, 0, stream>>>(xb, WcT, bc, Qb, Kb, Vt);

    // 3) flash attention: 8192 waves, 8 per block
    attn_flash<<<(BB * HH * (SS / 16)) / 8, 256, 0, stream>>>(Qb, Kb, Vt, Cc);

    // 4) output projection (M=8192, N=1024)
    gemm_out<<<dim3(MM / 32, DD / 256), 256, 0, stream>>>(Cc, WoT, bo, out);
}
